// AttentionLayer_51024211476506
// MI455X (gfx1250) — matrix operand order repaired and checked by reading, not compiled, not run on hardware
//
#include <hip/hip_runtime.h>

// ---------------------------------------------------------------------------
// Problem constants (from setup_inputs): B=2, Lq=Lk=2048, D=F=128, H=8
// ---------------------------------------------------------------------------
#define BB   2
#define LQ   2048
#define LK   2048
#define DD   128
#define FF   128
#define HH   8
#define TQ   128          // q rows per block (8 waves x 16 rows)
#define TK   64           // k tile
#define KHF  (HH * FF)    // 1024, K-dim of final projection

typedef __attribute__((ext_vector_type(16))) __bf16 v16bf;
typedef __attribute__((ext_vector_type(8)))  __bf16 v8bf;
typedef __attribute__((ext_vector_type(8)))  float  v8f;
typedef __attribute__((ext_vector_type(4)))  float  v4f;

static __device__ __forceinline__ v8f wmma_bf16(v16bf a, v16bf b, v8f c) {
    return __builtin_amdgcn_wmma_f32_16x16x32_bf16(
        /*neg_a=*/false, a, /*neg_b=*/false, b,
        /*c_mod=*/(short)0, c, /*reuse_a=*/false, /*reuse_b=*/false);
}

// ---------------------------------------------------------------------------
// Kernel 1: flash attention per (b, h, q-tile).
//   scores = (XQ * g[h] * 1/sqrt(D)) @ XK^T ; masked softmax ; P @ XV
//   Oh[b,q,h*F+f] = (P@XV)[q,f] * dv[h,f] / l[q]     (bf16 workspace)
// ---------------------------------------------------------------------------
__global__ __launch_bounds__(256)
void attn_flash_kernel(const float* __restrict__ XQ, const float* __restrict__ XK,
                       const float* __restrict__ XV, const int* __restrict__ mask,
                       const float* __restrict__ WQ, const float* __restrict__ WK,
                       const float* __restrict__ WV, __bf16* __restrict__ OhWs)
{
    __shared__ float g_s[DD];     // dq*dk*scale per d
    __shared__ float dv_s[FF];    // diag(W_V)
    alignas(32) __shared__ __bf16 Ks[TK][DD];     // K tile  [k][d]   16 KB
    alignas(32) __shared__ __bf16 VTs[FF][TK];    // V tileT [f][k]   16 KB
    alignas(32) __shared__ __bf16 Ps[8][16][TK];  // per-wave P patch 16 KB

    const int tid  = threadIdx.x;
    const int wave = tid >> 5;
    const int lane = tid & 31;
    const int hh   = lane >> 4;      // lane-half (0/1)
    const int l16  = lane & 15;

    const int nqt = LQ / TQ;         // 16
    const int b   = blockIdx.x / (HH * nqt);
    const int h   = (blockIdx.x / nqt) % HH;
    const int q0  = (blockIdx.x % nqt) * TQ;

    // ---- stage per-head diagonal scales -----------------------------------
    if (tid < DD) {
        const int d = tid;
        const float wq = WQ[((size_t)(h * DD) + d) * DD + d];
        const float wk = WK[((size_t)(h * DD) + d) * DD + d];
        g_s[d]  = wq * wk * 0.08838834764831845f;   // 1/sqrt(128)
        dv_s[d] = WV[((size_t)(h * FF) + d) * FF + d];
    }
    __syncthreads();

    // ---- Q-tilde rows: registers, A operand in contiguous k order (k = 16*half + e) -------------
    const int qrow = q0 + wave * 16 + l16;
    const float* xq = XQ + ((size_t)b * LQ + qrow) * DD;
    v16bf aq[4];
    #pragma unroll
    for (int c = 0; c < 4; ++c) {
        const int d0 = c * 32 + hh * 16;      // e0..7  -> K = d0..d0+7
        const int d1 = d0 + 8;                // e8..15 -> K = d1..d1+7
        #pragma unroll
        for (int e = 0; e < 8; ++e) {
            aq[c][e]     = (__bf16)(xq[d0 + e] * g_s[d0 + e]);
            aq[c][8 + e] = (__bf16)(xq[d1 + e] * g_s[d1 + e]);
        }
    }

    v8f acc[8];
    #pragma unroll
    for (int i = 0; i < 8; ++i)
        #pragma unroll
        for (int j = 0; j < 8; ++j) acc[i][j] = 0.0f;

    float mrow[8], lrow[8];
    #pragma unroll
    for (int v = 0; v < 8; ++v) { mrow[v] = -3.0e38f; lrow[v] = 0.0f; }

    const float* xk_base = XK + (size_t)b * LK * DD;
    const float* xv_base = XV + (size_t)b * LK * DD;
    const int*   mk_base = mask + (size_t)b * LQ * LK;

    for (int kt = 0; kt < LK; kt += TK) {
        __syncthreads();
        // ---- stage K tile (f32 -> bf16) -----------------------------------
        for (int i = tid; i < (TK * DD) / 4; i += 256) {
            const int idx = i << 2;
            const int kk = idx >> 7, dd = idx & 127;
            v4f t = *(const v4f*)(xk_base + (size_t)(kt + kk) * DD + dd);
            Ks[kk][dd + 0] = (__bf16)t.x; Ks[kk][dd + 1] = (__bf16)t.y;
            Ks[kk][dd + 2] = (__bf16)t.z; Ks[kk][dd + 3] = (__bf16)t.w;
        }
        // ---- stage V tile transposed --------------------------------------
        for (int i = tid; i < (TK * FF) / 4; i += 256) {
            const int idx = i << 2;
            const int kk = idx >> 7, ff = idx & 127;
            v4f t = *(const v4f*)(xv_base + (size_t)(kt + kk) * FF + ff);
            VTs[ff + 0][kk] = (__bf16)t.x; VTs[ff + 1][kk] = (__bf16)t.y;
            VTs[ff + 2][kk] = (__bf16)t.z; VTs[ff + 3][kk] = (__bf16)t.w;
        }
        if (kt + TK < LK) {   // global_prefetch_b8 for next tiles
            __builtin_prefetch(xk_base + (size_t)(kt + TK) * DD, 0, 1);
            __builtin_prefetch(xv_base + (size_t)(kt + TK) * FF, 0, 1);
        }
        __syncthreads();

        // ---- S = Qtilde @ K^T ---------------------------------------------
        // Batch B-operand loads (distinct regs) so WMMAs don't serialize on
        // s_wait_dscnt 0: per 32-wide d-chunk load 4 col-block operands, then
        // issue 4 independent WMMAs.
        v8f s[4];
        #pragma unroll
        for (int cb = 0; cb < 4; ++cb)
            #pragma unroll
            for (int j = 0; j < 8; ++j) s[cb][j] = 0.0f;

        #pragma unroll
        for (int c = 0; c < 4; ++c) {
            v16bf bk[4];
            #pragma unroll
            for (int cb = 0; cb < 4; ++cb)
                bk[cb] = *(const v16bf*)&Ks[cb * 16 + l16][c * 32 + hh * 16];
            #pragma unroll
            for (int cb = 0; cb < 4; ++cb)
                s[cb] = wmma_bf16(aq[c], bk[cb], s[cb]);
        }

        // ---- mask ---------------------------------------------------------
        #pragma unroll
        for (int cb = 0; cb < 4; ++cb) {
            #pragma unroll
            for (int v = 0; v < 8; ++v) {
                const int q = q0 + wave * 16 + hh * 8 + v;
                const int k = kt + cb * 16 + l16;
                if (mk_base[(size_t)q * LK + k] == 0) s[cb][v] = -3.0e38f;
            }
        }

        // ---- online softmax (row = vgpr + 8*half, 16 lanes per half) ------
        float alpha[8];
        #pragma unroll
        for (int v = 0; v < 8; ++v) {
            float mx = fmaxf(fmaxf(s[0][v], s[1][v]), fmaxf(s[2][v], s[3][v]));
            #pragma unroll
            for (int off = 1; off < 16; off <<= 1)
                mx = fmaxf(mx, __shfl_xor(mx, off, 32));
            const float mnew = fmaxf(mrow[v], mx);
            alpha[v] = __expf(mrow[v] - mnew);
            mrow[v]  = mnew;
        }
        #pragma unroll
        for (int v = 0; v < 8; ++v) {
            float ssum = 0.0f;
            #pragma unroll
            for (int cb = 0; cb < 4; ++cb) {
                const float p = __expf(s[cb][v] - mrow[v]);
                s[cb][v] = p;
                ssum += p;
            }
            #pragma unroll
            for (int off = 1; off < 16; off <<= 1)
                ssum += __shfl_xor(ssum, off, 32);
            lrow[v] = lrow[v] * alpha[v] + ssum;
            #pragma unroll
            for (int fb = 0; fb < 8; ++fb) acc[fb][v] *= alpha[v];
        }

        // ---- C-layout P -> LDS -> A-layout --------------------------------
        #pragma unroll
        for (int cb = 0; cb < 4; ++cb)
            #pragma unroll
            for (int v = 0; v < 8; ++v)
                Ps[wave][v + 8 * hh][cb * 16 + l16] = (__bf16)s[cb][v];

        v16bf ap[2];
        #pragma unroll
        for (int kc = 0; kc < 2; ++kc) {
            const __bf16* pp = &Ps[wave][l16][kc * 32 + hh * 16];
            v8bf lo = *(const v8bf*)(pp);
            v8bf hv = *(const v8bf*)(pp + 8);
            #pragma unroll
            for (int e = 0; e < 8; ++e) { ap[kc][e] = lo[e]; ap[kc][8 + e] = hv[e]; }
        }

        // ---- O += P @ V : batch 4 B operands (2 f-blocks x 2 kc) per group -
        #pragma unroll
        for (int fbg = 0; fbg < 4; ++fbg) {
            v16bf bv[4];
            #pragma unroll
            for (int j = 0; j < 2; ++j)
                #pragma unroll
                for (int kc = 0; kc < 2; ++kc)
                    bv[j * 2 + kc] = *(const v16bf*)
                        &VTs[(fbg * 2 + j) * 16 + l16][kc * 32 + hh * 16];
            #pragma unroll
            for (int j = 0; j < 2; ++j)
                #pragma unroll
                for (int kc = 0; kc < 2; ++kc)
                    acc[fbg * 2 + j] =
                        wmma_bf16(ap[kc], bv[j * 2 + kc], acc[fbg * 2 + j]);
        }
    }

    // ---- finalize: /l, *dv, store bf16 Oh [b][q][h*F+f] -------------------
    float inv[8];
    #pragma unroll
    for (int v = 0; v < 8; ++v) inv[v] = lrow[v] > 0.0f ? 1.0f / lrow[v] : 0.0f;

    #pragma unroll
    for (int fb = 0; fb < 8; ++fb) {
        const int f = fb * 16 + l16;
        const float dvf = dv_s[f];
        #pragma unroll
        for (int v = 0; v < 8; ++v) {
            const int q = q0 + wave * 16 + hh * 8 + v;
            OhWs[((size_t)(b * LQ + q)) * KHF + h * FF + f] =
                (__bf16)(acc[fb][v] * inv[v] * dvf);
        }
    }
}

// ---------------------------------------------------------------------------
// Kernel 2: out[m, f] = Oh[m, :] @ O[:, f]   (M=4096, N=128, K=1024), WMMA bf16
// ---------------------------------------------------------------------------
__global__ __launch_bounds__(256)
void proj_kernel(const __bf16* __restrict__ Oh, const float* __restrict__ O,
                 float* __restrict__ out)
{
    alignas(32) __shared__ __bf16 OTs[FF][TK];   // O^T tile [f][k], 16 KB

    const int tid  = threadIdx.x;
    const int wave = tid >> 5;
    const int lane = tid & 31;
    const int hh   = lane >> 4;
    const int l16  = lane & 15;

    const int row0 = blockIdx.x * 128;
    const int row  = row0 + wave * 16 + l16;

    v8f acc[8];
    #pragma unroll
    for (int i = 0; i < 8; ++i)
        #pragma unroll
        for (int j = 0; j < 8; ++j) acc[i][j] = 0.0f;

    for (int kt = 0; kt < KHF; kt += TK) {
        __syncthreads();
        for (int i = tid; i < (TK * FF) / 4; i += 256) {
            const int idx = i << 2;
            const int kk = idx >> 7, ff = idx & 127;
            v4f t = *(const v4f*)(O + (size_t)(kt + kk) * FF + ff);
            OTs[ff + 0][kk] = (__bf16)t.x; OTs[ff + 1][kk] = (__bf16)t.y;
            OTs[ff + 2][kk] = (__bf16)t.z; OTs[ff + 3][kk] = (__bf16)t.w;
        }
        if (kt + TK < KHF)
            __builtin_prefetch(O + (size_t)(kt + TK) * FF, 0, 1);
        __syncthreads();

        // A operand: bf16 workspace is already A-layout contiguous
        v16bf a[2];
        const __bf16* arow = Oh + (size_t)row * KHF + kt;
        #pragma unroll
        for (int kc = 0; kc < 2; ++kc) {
            v8bf lo = *(const v8bf*)(arow + kc * 32 + hh * 16);
            v8bf hv = *(const v8bf*)(arow + kc * 32 + hh * 16 + 8);
            #pragma unroll
            for (int e = 0; e < 8; ++e) { a[kc][e] = lo[e]; a[kc][8 + e] = hv[e]; }
        }

        #pragma unroll
        for (int fbg = 0; fbg < 4; ++fbg) {
            v16bf bo[4];
            #pragma unroll
            for (int j = 0; j < 2; ++j)
                #pragma unroll
                for (int kc = 0; kc < 2; ++kc)
                    bo[j * 2 + kc] = *(const v16bf*)
                        &OTs[(fbg * 2 + j) * 16 + l16][kc * 32 + hh * 16];
            #pragma unroll
            for (int j = 0; j < 2; ++j)
                #pragma unroll
                for (int kc = 0; kc < 2; ++kc)
                    acc[fbg * 2 + j] =
                        wmma_bf16(a[kc], bo[j * 2 + kc], acc[fbg * 2 + j]);
        }
    }

    #pragma unroll
    for (int fb = 0; fb < 8; ++fb) {
        const int f = fb * 16 + l16;
        #pragma unroll
        for (int v = 0; v < 8; ++v) {
            const int m = row0 + wave * 16 + hh * 8 + v;
            out[(size_t)m * FF + f] = acc[fb][v];
        }
    }
}

// ---------------------------------------------------------------------------
extern "C" void kernel_launch(void* const* d_in, const int* in_sizes, int n_in,
                              void* d_out, int out_size, void* d_ws, size_t ws_size,
                              hipStream_t stream) {
    (void)in_sizes; (void)n_in; (void)out_size; (void)ws_size;
    const float* XQ   = (const float*)d_in[0];
    const float* XK   = (const float*)d_in[1];
    const float* XV   = (const float*)d_in[2];
    const int*   mask = (const int*)d_in[3];
    const float* WQ   = (const float*)d_in[4];
    const float* WK   = (const float*)d_in[5];
    const float* WV   = (const float*)d_in[6];
    const float* O    = (const float*)d_in[7];
    __bf16* OhWs = (__bf16*)d_ws;            // needs B*Lq*H*F*2 = 8 MB scratch
    float*  out  = (float*)d_out;

    attn_flash_kernel<<<BB * HH * (LQ / TQ), 256, 0, stream>>>(
        XQ, XK, XV, mask, WQ, WK, WV, OhWs);
    proj_kernel<<<(BB * LQ) / TQ, 256, 0, stream>>>(OhWs, O, out);
}
